// KAN_layer_63333587746983
// MI455X (gfx1250) — matrix operand order repaired and checked by reading, not compiled, not run on hardware
//
#include <hip/hip_runtime.h>
#include <hip/hip_bf16.h>
#include <math.h>

// ---------------------------------------------------------------------------
// KAN layer on MI455X (gfx1250), restructured as a single f16 WMMA GEMM:
//   out[2048,256] = A[2048,3072] @ Wt[256,3072]^T   (f32 accumulate)
// where A packs [11 cubic-B-spline basis values + silu] per (b,i) and Wt
// pre-folds mask*scale_spline*control_points and mask*scale_base.
// ---------------------------------------------------------------------------

#define KDEG   3
#define NUMIV  8
#define INDIM  256
#define OUTDIM 256
#define BATCH  2048
#define G      (NUMIV + 1 + 2 * KDEG)   // 15 knots per input
#define NCOEF  (G - (KDEG + 1))         // 11 basis functions
#define LPACK  (NCOEF + 1)              // 12 channels per input (11 basis + silu)
#define KD     (INDIM * LPACK)          // 3072 GEMM K dimension

typedef __attribute__((ext_vector_type(4)))  _Float16 v4h;
typedef __attribute__((ext_vector_type(8)))  _Float16 v8h;
typedef __attribute__((ext_vector_type(16))) _Float16 v16h;
typedef __attribute__((ext_vector_type(8)))  float    v8f;

// ---------------------------------------------------------------------------
// Phase 1: per-(b,i) Cox-de Boor basis (degree 3) + silu -> A[b, i*12 + l]
// One thread per (b,i); 12 f16 results stored as 3x v4h (8-byte aligned).
// ---------------------------------------------------------------------------
__global__ __launch_bounds__(256)
void kan_basis_kernel(const float* __restrict__ x,
                      const float* __restrict__ knots,
                      _Float16* __restrict__ A) {
    const int t = blockIdx.x * blockDim.x + threadIdx.x;   // [0, BATCH*INDIM)
    const int i = t & (INDIM - 1);
    const float xv = x[t];

    float tk[G];
#pragma unroll
    for (int j = 0; j < G; ++j) tk[j] = knots[i * G + j];

    // degree-0 indicator basis
    float Bv[G - 1];
#pragma unroll
    for (int j = 0; j < G - 1; ++j)
        Bv[j] = (xv >= tk[j] && xv < tk[j + 1]) ? 1.0f : 0.0f;

    // Cox-de Boor recursion up to degree 3
#pragma unroll
    for (int p = 1; p <= KDEG; ++p) {
#pragma unroll
        for (int j = 0; j < G - 1 - p; ++j) {
            const float left  = (xv - tk[j]) / (tk[j + p] - tk[j]);
            const float right = (tk[j + p + 1] - xv) / (tk[j + p + 1] - tk[j + 1]);
            Bv[j] = left * Bv[j] + right * Bv[j + 1];
        }
    }

    const float silu = xv / (1.0f + __expf(-xv));

    _Float16 w[LPACK];
#pragma unroll
    for (int l = 0; l < NCOEF; ++l) w[l] = (_Float16)Bv[l];
    w[NCOEF] = (_Float16)silu;

    v4h* dst = (v4h*)(A + (size_t)t * LPACK);
    dst[0] = v4h{w[0], w[1], w[2],  w[3]};
    dst[1] = v4h{w[4], w[5], w[6],  w[7]};
    dst[2] = v4h{w[8], w[9], w[10], w[11]};
}

// ---------------------------------------------------------------------------
// Phase 2: fold scales/mask into f16 weights, stored transposed:
//   Wt[o, i*12 + l] = mask[i,o]*scale_spline[i,o]*cp[i,o,l]   (l < 11)
//   Wt[o, i*12 +11] = mask[i,o]*scale_base[i,o]
// One thread per (i,o) pair, o fastest for coalesced scale/mask/cp reads.
// ---------------------------------------------------------------------------
__global__ __launch_bounds__(256)
void kan_weights_kernel(const float* __restrict__ cp,
                        const float* __restrict__ scale_base,
                        const float* __restrict__ scale_spline,
                        const float* __restrict__ mask,
                        _Float16* __restrict__ Wt) {
    const int t = blockIdx.x * blockDim.x + threadIdx.x;   // [0, INDIM*OUTDIM)
    const int o = t & (OUTDIM - 1);
    const int i = t >> 8;
    const int io = i * OUTDIM + o;

    const float m   = mask[io];
    const float vss = scale_spline[io] * m;
    const float vsb = scale_base[io]   * m;
    const float* c  = cp + (size_t)io * NCOEF;

    _Float16 w[LPACK];
#pragma unroll
    for (int l = 0; l < NCOEF; ++l) w[l] = (_Float16)(vss * c[l]);
    w[NCOEF] = (_Float16)vsb;

    v4h* dst = (v4h*)(Wt + (size_t)o * KD + (size_t)i * LPACK);
    dst[0] = v4h{w[0], w[1], w[2],  w[3]};
    dst[1] = v4h{w[4], w[5], w[6],  w[7]};
    dst[2] = v4h{w[8], w[9], w[10], w[11]};
}

// ---------------------------------------------------------------------------
// Phase 3: C[2048,256] = A[2048,3072] x Wt[256,3072]^T via
// v_wmma_f32_16x16x32_f16. 256-thread block = 8 wave32 waves; each wave owns
// a 16(M) x 64(N) output tile -> block tile 128 x 64; grid (16, 4).
// K loop: 96 steps of 32, 4 WMMAs per step per wave.
//
// Software-pipelined with a rolled loop (#pragma unroll 1): each iteration
// issues the 10 b128 loads for step k+1 BEFORE the 4 WMMAs of step k, so the
// steady-state wait before the WMMAs only drains the PREVIOUS step's loads
// while the next step's 10 loads stay in flight behind the matrix ops.
//
// A fragment: lane<16 -> row M=lane, elements 0..15 = K 0..15;
// lane>=16 -> same row (lane&15), K 16..31 (contiguous, matches B).
// B fragment: lane owns output column N=(lane&15); elements
// are a contiguous run of K from the row-major Wt row, upper lane-half
// offset +16 in K (mirrors the documented SWMMAC dense-B feed).
// ---------------------------------------------------------------------------
__device__ __forceinline__ v16h load_a_frag(const _Float16* Ap, int kk) {
    v8h lo = *(const v8h*)(Ap + kk);
    v8h hi = *(const v8h*)(Ap + kk + 8);
    return __builtin_shufflevector(lo, hi,
            0, 1, 2, 3, 4, 5, 6, 7, 8, 9, 10, 11, 12, 13, 14, 15);
}

__global__ __launch_bounds__(256)
void kan_gemm_kernel(const _Float16* __restrict__ A,
                     const _Float16* __restrict__ Wt,
                     float* __restrict__ C) {
    const int lane = threadIdx.x & 31;
    const int wave = threadIdx.x >> 5;
    const int Moff = blockIdx.x * 128 + wave * 16;
    const int Noff = blockIdx.y * 64;

    const int mrow = Moff + (lane & 15);
    const int aK   = (lane & 16) ? 16 : 0;   // A half-K element offset
    const int bK   = (lane & 16) ? 16 : 0;   // B half-K element offset

    const _Float16* Ap = A  + (size_t)mrow * KD + aK;
    const _Float16* B0 = Wt + (size_t)(Noff + (lane & 15)) * KD + bK;
    const _Float16* B1 = B0 + (size_t)16 * KD;
    const _Float16* B2 = B0 + (size_t)32 * KD;
    const _Float16* B3 = B0 + (size_t)48 * KD;

    v8f acc0 = {}; v8f acc1 = {}; v8f acc2 = {}; v8f acc3 = {};

    // ---- pipeline prologue: fragments for kk = 0 ----
    v16h a  = load_a_frag(Ap, 0);
    v16h b0 = *(const v16h*)(B0);
    v16h b1 = *(const v16h*)(B1);
    v16h b2 = *(const v16h*)(B2);
    v16h b3 = *(const v16h*)(B3);

    // Rolled loop: keep the register rotation (and thus the load/WMMA
    // overlap) intact instead of letting full unroll re-serialize it.
#pragma unroll 1
    for (int kk = 0; kk < KD - 32; kk += 32) {
        const int kn = kk + 32;
        // prefetch one step further into GL2 (shared across waves/blocks)
        __builtin_prefetch(Ap + kn + 32, 0, 1);
        __builtin_prefetch(B0 + kn + 32, 0, 1);

        // issue next step's loads first (overlap with this step's WMMAs)
        v16h an  = load_a_frag(Ap, kn);
        v16h b0n = *(const v16h*)(B0 + kn);
        v16h b1n = *(const v16h*)(B1 + kn);
        v16h b2n = *(const v16h*)(B2 + kn);
        v16h b3n = *(const v16h*)(B3 + kn);

        acc0 = __builtin_amdgcn_wmma_f32_16x16x32_f16(false, a, false, b0,
                                                      (short)0, acc0, false, false);
        acc1 = __builtin_amdgcn_wmma_f32_16x16x32_f16(false, a, false, b1,
                                                      (short)0, acc1, false, false);
        acc2 = __builtin_amdgcn_wmma_f32_16x16x32_f16(false, a, false, b2,
                                                      (short)0, acc2, false, false);
        acc3 = __builtin_amdgcn_wmma_f32_16x16x32_f16(false, a, false, b3,
                                                      (short)0, acc3, false, false);

        a = an; b0 = b0n; b1 = b1n; b2 = b2n; b3 = b3n;
    }

    // ---- pipeline epilogue: last K step ----
    acc0 = __builtin_amdgcn_wmma_f32_16x16x32_f16(false, a, false, b0,
                                                  (short)0, acc0, false, false);
    acc1 = __builtin_amdgcn_wmma_f32_16x16x32_f16(false, a, false, b1,
                                                  (short)0, acc1, false, false);
    acc2 = __builtin_amdgcn_wmma_f32_16x16x32_f16(false, a, false, b2,
                                                  (short)0, acc2, false, false);
    acc3 = __builtin_amdgcn_wmma_f32_16x16x32_f16(false, a, false, b3,
                                                  (short)0, acc3, false, false);

    // C/D 16x16 f32 layout (ISA 7.12.2): VGPR v, lanes 0-15 -> M=v,
    // lanes 16-31 -> M=v+8; column N = lane&15.
    const int rbase = Moff + ((lane & 16) ? 8 : 0);
    const int col   = Noff + (lane & 15);
#pragma unroll
    for (int v = 0; v < 8; ++v) {
        float* row = C + (size_t)(rbase + v) * OUTDIM + col;
        row[0]  = acc0[v];
        row[16] = acc1[v];
        row[32] = acc2[v];
        row[48] = acc3[v];
    }
}

// ---------------------------------------------------------------------------
// Launch: three kernels on `stream` (implicit ordering on the stream).
// Workspace layout: A (2048*3072 f16 = 12.58 MB) then Wt (256*3072 f16).
// ---------------------------------------------------------------------------
extern "C" void kernel_launch(void* const* d_in, const int* in_sizes, int n_in,
                              void* d_out, int out_size, void* d_ws, size_t ws_size,
                              hipStream_t stream) {
    const float* x            = (const float*)d_in[0];   // [2048, 256]
    const float* knots        = (const float*)d_in[1];   // [256, 15]
    const float* cp           = (const float*)d_in[2];   // [256, 256, 11]
    const float* scale_base   = (const float*)d_in[3];   // [256, 256]
    const float* scale_spline = (const float*)d_in[4];   // [256, 256]
    const float* mask         = (const float*)d_in[5];   // [256, 256]
    float* out = (float*)d_out;                          // [2048, 256]

    _Float16* Abuf = (_Float16*)d_ws;
    _Float16* Wt   = (_Float16*)((char*)d_ws + (size_t)BATCH * KD * sizeof(_Float16));

    kan_basis_kernel<<<(BATCH * INDIM) / 256, 256, 0, stream>>>(x, knots, Abuf);
    kan_weights_kernel<<<(INDIM * OUTDIM) / 256, 256, 0, stream>>>(
        cp, scale_base, scale_spline, mask, Wt);

    dim3 grid(BATCH / 128, OUTDIM / 64);
    kan_gemm_kernel<<<grid, 256, 0, stream>>>(Abuf, Wt, out);
}
